// SimplerDimplerModel_24326694764806
// MI455X (gfx1250) — matrix operand order repaired and checked by reading, not compiled, not run on hardware
//
#include <hip/hip_runtime.h>
#include <hip/hip_bf16.h>

// ---------------------------------------------------------------------------
// Types for CDNA5 WMMA (gfx1250, wave32): D(16x16 f32) = A(16x32 bf16) x B(32x16 bf16) + C
// ---------------------------------------------------------------------------
typedef __attribute__((ext_vector_type(16))) __bf16 v16bf;
typedef __attribute__((ext_vector_type(8)))  float  v8f;
typedef __attribute__((ext_vector_type(4)))  unsigned int u32x4;
typedef __attribute__((ext_vector_type(8)))  unsigned int u32x8;
typedef int v4i __attribute__((__vector_size__(16)));   // matches async-LDS builtin proto

union Frag {
  v16bf v;
  uint4 q[2];
};

__device__ __forceinline__ v8f wmma_bf16(const Frag& a, const Frag& b, v8f c) {
  // 8 args: (neg_a, A, neg_b, B, c_mod, C, reuse_a, reuse_b)
  return __builtin_amdgcn_wmma_f32_16x16x32_bf16(false, a.v, false, b.v,
                                                 (short)0, c, false, false);
}

__device__ __forceinline__ v8f v8f_zero() {
  v8f z = {0.f, 0.f, 0.f, 0.f, 0.f, 0.f, 0.f, 0.f};
  return z;
}

// Async global->LDS staging (ASYNCcnt path) with compile-safe fallback.
__device__ __forceinline__ void copy16_g2lds(const __bf16* gsrc, __bf16* ldst) {
#if __has_builtin(__builtin_amdgcn_global_load_async_to_lds_b128)
  __builtin_amdgcn_global_load_async_to_lds_b128(
      (__attribute__((address_space(1))) v4i*)gsrc,
      (__attribute__((address_space(3))) v4i*)ldst, 0, 0);
#else
  *(uint4*)ldst = *(const uint4*)gsrc;
#endif
}

__device__ __forceinline__ void wait_async_copies() {
#if __has_builtin(__builtin_amdgcn_global_load_async_to_lds_b128)
#if __has_builtin(__builtin_amdgcn_s_wait_asynccnt)
  __builtin_amdgcn_s_wait_asynccnt(0);
#else
  asm volatile("s_wait_asynccnt 0" ::: "memory");
#endif
#endif
}

// ---------------------------------------------------------------------------
// Elementwise: f32 -> bf16 weight conversion
// ---------------------------------------------------------------------------
__global__ __launch_bounds__(256) void f2bf_kernel(const float* __restrict__ in,
                                                   __bf16* __restrict__ out,
                                                   size_t n) {
  size_t i = (size_t)blockIdx.x * blockDim.x + threadIdx.x;
  if (i < n) out[i] = (__bf16)in[i];
}

// ---------------------------------------------------------------------------
// Embedding gather: xf = embeds[X], xb = bf16(xf)
// ---------------------------------------------------------------------------
__global__ __launch_bounds__(256) void embed_kernel(const int* __restrict__ X,
                                                    const float* __restrict__ emb,
                                                    float* __restrict__ xf,
                                                    __bf16* __restrict__ xb,
                                                    int BN, int E) {
  size_t i = (size_t)blockIdx.x * blockDim.x + threadIdx.x;
  if (i >= (size_t)BN * E) return;
  int tok = X[i / E];
  float v = emb[(size_t)tok * E + (i % E)];
  xf[i] = v;
  xb[i] = (__bf16)v;
}

// ---------------------------------------------------------------------------
// Tiled WMMA GEMM:  C[M,Nout] = A[M,K](bf16) * W[Nout,K]^T(bf16) + bias
// Block: 256 threads = 8 waves, 128x128 tile, K-step 64.
// Wave layout 2x4; each wave computes 64x32 (4 Mtiles x 2 Ntiles).
// Requires: M%128==0, Nout%128==0, K%64==0 (true for all calls here).
// ---------------------------------------------------------------------------
template <bool OUT_BF16, bool GELU>
__global__ __launch_bounds__(256) void gemm_bf16_kernel(
    const __bf16* __restrict__ A, const __bf16* __restrict__ W,
    const float* __restrict__ bias, void* __restrict__ Cout,
    int M, int Nout, int K) {
  __shared__ __bf16 lA[128][72];  // row stride 144B (multiple of 16B)
  __shared__ __bf16 lB[128][72];

  const int tid  = threadIdx.x;
  const int lane = tid & 31;
  const int wave = tid >> 5;
  const int wm   = (wave >> 2) * 64;   // 0 / 64
  const int wn   = (wave & 3) * 32;    // 0..96
  const int bm   = blockIdx.y * 128;
  const int bn   = blockIdx.x * 128;
  const int arow = lane & 15;
  const int chi  = lane >> 4;          // lane half
  const int kbl  = chi * 8;            // A-operand K base per half

  v8f acc[4][2];
#pragma unroll
  for (int rt = 0; rt < 4; ++rt)
#pragma unroll
    for (int nt = 0; nt < 2; ++nt) acc[rt][nt] = v8f_zero();

  for (int k0 = 0; k0 < K; k0 += 64) {
    // Cooperative load of 128x64-half A-tile and W-tile:
    // 1024 16B chunks per tile, 4 chunks per thread per tile.
#pragma unroll
    for (int c = 0; c < 4; ++c) {
      int chunk = tid * 4 + c;        // 0..1023
      int row   = chunk >> 3;         // 0..127
      int col   = (chunk & 7) * 8;    // halves: 0,8,...,56
      const __bf16* ap = &A[(size_t)(bm + row) * K + k0 + col];
      const __bf16* wp = &W[(size_t)(bn + row) * K + k0 + col];
      copy16_g2lds(ap, &lA[row][col]);
      copy16_g2lds(wp, &lB[row][col]);
      if (k0 + 64 < K) {              // prefetch next K-slab (global_prefetch_b8)
        __builtin_prefetch(ap + 64, 0, 1);
        __builtin_prefetch(wp + 64, 0, 1);
      }
    }
    wait_async_copies();
    __syncthreads();

#pragma unroll
    for (int kf = 0; kf < 2; ++kf) {
      // B operand: lane = output column; K = {kbl..kbl+7, kbl+16..kbl+23} per lane half
      Frag bfr[2];
#pragma unroll
      for (int nt = 0; nt < 2; ++nt) {
        int r    = wn + nt * 16 + arow;
        int boff = kf * 32 + chi * 8;
        bfr[nt].q[0] = *(const uint4*)&lB[r][boff];
        bfr[nt].q[1] = *(const uint4*)&lB[r][boff + 16];
      }
#pragma unroll
      for (int rt = 0; rt < 4; ++rt) {
        // A operand: lane = M row; K = {kbl..kbl+7, kbl+16..kbl+23}
        Frag afr;
        int r = wm + rt * 16 + arow;
        afr.q[0] = *(const uint4*)&lA[r][kf * 32 + kbl];
        afr.q[1] = *(const uint4*)&lA[r][kf * 32 + kbl + 16];
#pragma unroll
        for (int nt = 0; nt < 2; ++nt)
          acc[rt][nt] = wmma_bf16(afr, bfr[nt], acc[rt][nt]);
      }
    }
    __syncthreads();
  }

  // Epilogue: D layout -> row = r + 8*chi, col = arow
#pragma unroll
  for (int nt = 0; nt < 2; ++nt) {
    int col  = bn + wn + nt * 16 + arow;
    float bv = bias[col];
#pragma unroll
    for (int rt = 0; rt < 4; ++rt) {
#pragma unroll
      for (int r = 0; r < 8; ++r) {
        int row = bm + wm + rt * 16 + r + 8 * chi;
        float v = acc[rt][nt][r] + bv;
        if (GELU) v = 0.5f * v * (1.0f + erff(v * 0.70710678118654752f));
        size_t off = (size_t)row * Nout + col;
        if (OUT_BF16) ((__bf16*)Cout)[off] = (__bf16)v;
        else          ((float*)Cout)[off]  = v;
      }
    }
  }
}

// ---------------------------------------------------------------------------
// Flash attention with ALiBi + causal mask, Dh=64.
// Grid: (N/128, B*H). Block: 128 threads = 4 waves; each wave owns 32 q-rows.
// qkv layout: [B, N, 3E] bf16 (q at +0, k at +E, v at +2E), head h at +h*64.
// K tile is fetched by the Tensor Data Mover (tensor_load_to_lds, TENSORcnt).
// Output: ob [B, N, E] bf16.
// ---------------------------------------------------------------------------
__global__ __launch_bounds__(128) void flash_attn_kernel(
    const __bf16* __restrict__ qkv, __bf16* __restrict__ ob,
    int B, int N, int E, int H) {
  const int Dh = 64;
  __shared__ __bf16 lK[32][72];       // 32 keys x 64 dims (stride 144B, TDM-padded)
  __shared__ __bf16 lV[64][40];       // V transposed: [d][k] (stride 80B)
  __shared__ __bf16 lP[4][32][40];    // per-wave P tile, 32x32 (stride 80B)

  const int tid  = threadIdx.x;
  const int lane = tid & 31;
  const int wave = tid >> 5;
  const int bh   = blockIdx.y;
  const int b    = bh / H;
  const int h    = bh % H;
  const int qb   = blockIdx.x * 128;
  const int wq   = qb + wave * 32;    // this wave's first q row
  const int arow = lane & 15;
  const int chi  = lane >> 4;
  const int kbl  = chi * 8;
  const float slope = exp2f(-0.5f * (float)(h + 1));  // (2^(-8/16))^(h+1)
  const float scale = 0.125f;                          // 1/sqrt(64)
  const size_t rs = (size_t)3 * E;                     // qkv row stride (halves)

  // Q fragments (held in registers for the whole pass)
  Frag aq[2][2];
#pragma unroll
  for (int rt = 0; rt < 2; ++rt) {
    int row = wq + rt * 16 + arow;
    const __bf16* qp = qkv + (size_t)(b * N + row) * rs + h * Dh;
#pragma unroll
    for (int kf = 0; kf < 2; ++kf) {
      aq[rt][kf].q[0] = *(const uint4*)(qp + kf * 32 + kbl);
      aq[rt][kf].q[1] = *(const uint4*)(qp + kf * 32 + kbl + 16);
    }
  }

  v8f oacc[2][4];
  float mr[2][8], lr[2][8];
#pragma unroll
  for (int rt = 0; rt < 2; ++rt) {
#pragma unroll
    for (int dt = 0; dt < 4; ++dt) oacc[rt][dt] = v8f_zero();
#pragma unroll
    for (int r = 0; r < 8; ++r) { mr[rt][r] = -1e30f; lr[rt][r] = 0.f; }
  }

  const int kTiles = (qb + 128) / 32;   // causal extent of this q-block
  for (int kt = 0; kt < kTiles; ++kt) {
    const int kb = kt * 32;

    // ---- K tile (32 keys x 64 dims) via Tensor Data Mover (wave 0 issues) ----
    if (wave == 0) {
      uint64_t ga = (uint64_t)(uintptr_t)(qkv + (size_t)(b * N + kb) * rs + E + h * Dh);
      uint32_t la = (uint32_t)(uintptr_t)&lK[0][0];
      u32x4 g0;
      g0[0] = 1u;                                   // count=1, user descriptor
      g0[1] = la;                                   // lds_addr (bytes)
      g0[2] = (uint32_t)ga;                         // global_addr[31:0]
      g0[3] = (uint32_t)((ga >> 32) & 0x01FFFFFFu)  // global_addr[56:32]
              | (2u << 30);                         // type = 2 ("image")
      u32x8 g1;
      g1[0] = (1u << 16)       // data_size = 1 -> 2 bytes
            | (1u << 20)       // pad_enable: pad LDS rows
            | (4u << 22)       // pad_interval code 4 -> every 32 DWORDs (128B row)
            | (3u << 25);      // pad_amount code 3 -> 4 DWORDs (16B) -> stride 72 halves
      g1[1] = 64u << 16;       // tensor_dim0 = 64 (bits 79:48 low part)
      g1[2] = 32u << 16;       // tensor_dim1 = 32 (bits 111:80 low part)
      g1[3] = 64u << 16;       // tile_dim0 = 64 (bits 127:112)
      g1[4] = 32u;             // tile_dim1 = 32; tile_dim2 = 0
      g1[5] = 3u * (uint32_t)E;// tensor_dim0_stride = 3072 (low 32)
      g1[6] = 0u;              // stride0 high / stride1 low
      g1[7] = 0u;              // stride1 high
      asm volatile("tensor_load_to_lds %0, %1" :: "s"(g0), "s"(g1) : "memory");
#if __has_builtin(__builtin_amdgcn_s_wait_tensorcnt)
      __builtin_amdgcn_s_wait_tensorcnt(0);
#else
      asm volatile("s_wait_tensorcnt 0" ::: "memory");
#endif
    }
    // ---- V tile (32x64), loaded and stored transposed into lV[d][k] ----
    {
      int row = tid >> 2;             // 0..31
      int c0  = (tid & 3) * 16;       // 0,16,32,48
      const __bf16* vp = qkv + (size_t)(b * N + kb + row) * rs + 2 * E + h * Dh + c0;
      union { uint4 q[2]; __bf16 e[16]; } vv;
      vv.q[0] = *(const uint4*)vp;
      vv.q[1] = *(const uint4*)(vp + 8);
#pragma unroll
      for (int j = 0; j < 16; ++j) lV[c0 + j][row] = vv.e[j];
    }
    __syncthreads();

    if (kb <= wq + 31) {   // wave participates (not fully masked)
      // S = Q K^T : 2x2 f32 tiles, accumulate over Dh=64 (2 k-frags)
      v8f s[2][2];
#pragma unroll
      for (int rt = 0; rt < 2; ++rt)
#pragma unroll
        for (int nt = 0; nt < 2; ++nt) s[rt][nt] = v8f_zero();
#pragma unroll
      for (int kf = 0; kf < 2; ++kf) {
        Frag bk[2];
#pragma unroll
        for (int nt = 0; nt < 2; ++nt) {
          int r    = nt * 16 + arow;
          int boff = kf * 32 + chi * 8;
          bk[nt].q[0] = *(const uint4*)&lK[r][boff];
          bk[nt].q[1] = *(const uint4*)&lK[r][boff + 16];
        }
#pragma unroll
        for (int rt = 0; rt < 2; ++rt)
#pragma unroll
          for (int nt = 0; nt < 2; ++nt)
            s[rt][nt] = wmma_bf16(aq[rt][kf], bk[nt], s[rt][nt]);
      }

      // Online softmax: scale + ALiBi + causal, row-max / row-sum via 16-lane shuffles
#pragma unroll
      for (int rt = 0; rt < 2; ++rt) {
#pragma unroll
        for (int r = 0; r < 8; ++r) {
          int rowa = wq + rt * 16 + r + 8 * chi;
          int c0a  = kb + arow;
          int c1a  = kb + 16 + arow;
          float v0 = s[rt][0][r] * scale +
                     ((c0a <= rowa) ? slope * (float)(c0a - rowa) : -1e30f);
          float v1 = s[rt][1][r] * scale +
                     ((c1a <= rowa) ? slope * (float)(c1a - rowa) : -1e30f);
          float lm = fmaxf(v0, v1);
#pragma unroll
          for (int m = 1; m < 16; m <<= 1) lm = fmaxf(lm, __shfl_xor(lm, m, 16));
          float mnew = fmaxf(mr[rt][r], lm);
          float fsc  = __expf(mr[rt][r] - mnew);
          mr[rt][r]  = mnew;
          float p0 = __expf(v0 - mnew);
          float p1 = __expf(v1 - mnew);
          float ps = p0 + p1;
#pragma unroll
          for (int m = 1; m < 16; m <<= 1) ps += __shfl_xor(ps, m, 16);
          lr[rt][r] = lr[rt][r] * fsc + ps;
#pragma unroll
          for (int dt = 0; dt < 4; ++dt) oacc[rt][dt][r] *= fsc;
          int prow = rt * 16 + r + 8 * chi;
          lP[wave][prow][arow]      = (__bf16)p0;
          lP[wave][prow][16 + arow] = (__bf16)p1;
        }
      }
      // Same-wave LDS RAW before re-reading P as WMMA operand
      asm volatile("s_wait_dscnt 0" ::: "memory");

      // O += P(32x32) * V(32x64): exact 16x16x32 shape
      Frag pa[2];
#pragma unroll
      for (int rt = 0; rt < 2; ++rt) {
        int r = rt * 16 + arow;
        pa[rt].q[0] = *(const uint4*)&lP[wave][r][kbl];
        pa[rt].q[1] = *(const uint4*)&lP[wave][r][kbl + 16];
      }
#pragma unroll
      for (int dt = 0; dt < 4; ++dt) {
        Frag bv;
        int d    = dt * 16 + arow;
        int boff = chi * 8;
        bv.q[0] = *(const uint4*)&lV[d][boff];
        bv.q[1] = *(const uint4*)&lV[d][boff + 16];
#pragma unroll
        for (int rt = 0; rt < 2; ++rt)
          oacc[rt][dt] = wmma_bf16(pa[rt], bv, oacc[rt][dt]);
      }
    }
    __syncthreads();
  }

  // Epilogue: O /= l, store bf16 into [B,N,E] at head offset
#pragma unroll
  for (int rt = 0; rt < 2; ++rt) {
#pragma unroll
    for (int dt = 0; dt < 4; ++dt) {
#pragma unroll
      for (int r = 0; r < 8; ++r) {
        int rowa = wq + rt * 16 + r + 8 * chi;
        int col  = h * Dh + dt * 16 + arow;
        float v  = oacc[rt][dt][r] / lr[rt][r];
        ob[(size_t)(b * N + rowa) * E + col] = (__bf16)v;
      }
    }
  }
}

// ---------------------------------------------------------------------------
// Fused residual add + LayerNorm; outputs f32 residual stream and bf16 copy.
// One block per row. xo may alias res.
// ---------------------------------------------------------------------------
__global__ __launch_bounds__(256) void add_ln_kernel(
    const float* __restrict__ res, const float* __restrict__ y,
    const float* __restrict__ g, const float* __restrict__ bta,
    float* __restrict__ xo, __bf16* __restrict__ xb, int E) {
  __shared__ float sred[256];
  const int row = blockIdx.x;
  const int tid = threadIdx.x;
  const float* rp = res + (size_t)row * E;
  const float* yp = y + (size_t)row * E;

  float sum = 0.f;
  for (int e = tid; e < E; e += 256) sum += rp[e] + yp[e];
  sred[tid] = sum;
  __syncthreads();
  for (int s = 128; s > 0; s >>= 1) {
    if (tid < s) sred[tid] += sred[tid + s];
    __syncthreads();
  }
  float mean = sred[0] / (float)E;
  __syncthreads();

  float vs = 0.f;
  for (int e = tid; e < E; e += 256) {
    float d = rp[e] + yp[e] - mean;
    vs += d * d;
  }
  sred[tid] = vs;
  __syncthreads();
  for (int s = 128; s > 0; s >>= 1) {
    if (tid < s) sred[tid] += sred[tid + s];
    __syncthreads();
  }
  float rstd = rsqrtf(sred[0] / (float)E + 1e-5f);
  __syncthreads();

  for (int e = tid; e < E; e += 256) {
    float v = (rp[e] + yp[e] - mean) * rstd * g[e] + bta[e];
    xo[(size_t)row * E + e] = v;
    xb[(size_t)row * E + e] = (__bf16)v;
  }
}

// ---------------------------------------------------------------------------
// Host-side launcher
// ---------------------------------------------------------------------------
extern "C" void kernel_launch(void* const* d_in, const int* in_sizes, int n_in,
                              void* d_out, int out_size, void* d_ws, size_t ws_size,
                              hipStream_t stream) {
  (void)in_sizes; (void)n_in; (void)out_size; (void)ws_size;
  const int B = 2, N = 2048, E = 1024, H = 16, F = 4096, L = 4, V = 32000;
  const int M = B * N;  // 4096 rows

  const int*   X        = (const int*)d_in[0];
  const float* embeds   = (const float*)d_in[2];
  const float* inW      = (const float*)d_in[3];
  const float* inB      = (const float*)d_in[4];
  const float* outW     = (const float*)d_in[5];
  const float* outB     = (const float*)d_in[6];
  const float* ln1w     = (const float*)d_in[7];
  const float* ln1b     = (const float*)d_in[8];
  const float* ln2w     = (const float*)d_in[9];
  const float* ln2b     = (const float*)d_in[10];
  const float* l1W      = (const float*)d_in[11];
  const float* l1B      = (const float*)d_in[12];
  const float* l2W      = (const float*)d_in[13];
  const float* l2B      = (const float*)d_in[14];
  const float* headW    = (const float*)d_in[15];
  const float* headB    = (const float*)d_in[16];
  float* out            = (float*)d_out;

  // Workspace layout
  size_t off = 0;
  auto alloc = [&](size_t bytes) {
    void* p = (char*)d_ws + off;
    off += (bytes + 255) & ~(size_t)255;
    return p;
  };
  float*  xf     = (float*)alloc((size_t)M * E * 4);
  float*  tmp    = (float*)alloc((size_t)M * E * 4);
  __bf16* xb     = (__bf16*)alloc((size_t)M * E * 2);
  __bf16* ob     = (__bf16*)alloc((size_t)M * E * 2);
  __bf16* qkvb   = (__bf16*)alloc((size_t)M * 3 * E * 2);
  __bf16* h1     = (__bf16*)alloc((size_t)M * F * 2);
  __bf16* winb   = (__bf16*)alloc((size_t)L * 3 * E * E * 2);
  __bf16* woutb  = (__bf16*)alloc((size_t)L * E * E * 2);
  __bf16* wl1b   = (__bf16*)alloc((size_t)L * F * E * 2);
  __bf16* wl2b   = (__bf16*)alloc((size_t)L * E * F * 2);
  __bf16* wheadb = (__bf16*)alloc((size_t)V * E * 2);

  auto cvt = [&](const float* src, __bf16* dst, size_t n) {
    f2bf_kernel<<<(unsigned)((n + 255) / 256), 256, 0, stream>>>(src, dst, n);
  };
  cvt(inW,   winb,   (size_t)L * 3 * E * E);
  cvt(outW,  woutb,  (size_t)L * E * E);
  cvt(l1W,   wl1b,   (size_t)L * F * E);
  cvt(l2W,   wl2b,   (size_t)L * E * F);
  cvt(headW, wheadb, (size_t)V * E);

  {
    size_t n = (size_t)M * E;
    embed_kernel<<<(unsigned)((n + 255) / 256), 256, 0, stream>>>(X, embeds, xf, xb, M, E);
  }

  for (int l = 0; l < L; ++l) {
    // qkv = x @ Wqkv^T + bqkv  -> bf16 [M, 3E]
    gemm_bf16_kernel<true, false><<<dim3(3 * E / 128, M / 128), 256, 0, stream>>>(
        xb, winb + (size_t)l * 3 * E * E, inB + (size_t)l * 3 * E, qkvb, M, 3 * E, E);
    // attention -> ob bf16 [M, E]
    flash_attn_kernel<<<dim3(N / 128, B * H), 128, 0, stream>>>(qkvb, ob, B, N, E, H);
    // o @ Wo^T + bo -> tmp f32; x = LN(x + tmp)
    gemm_bf16_kernel<false, false><<<dim3(E / 128, M / 128), 256, 0, stream>>>(
        ob, woutb + (size_t)l * E * E, outB + (size_t)l * E, tmp, M, E, E);
    add_ln_kernel<<<M, 256, 0, stream>>>(xf, tmp, ln1w + (size_t)l * E, ln1b + (size_t)l * E,
                                         xf, xb, E);
    // h = gelu(x @ W1^T + c1) -> bf16 [M, F]
    gemm_bf16_kernel<true, true><<<dim3(F / 128, M / 128), 256, 0, stream>>>(
        xb, wl1b + (size_t)l * F * E, l1B + (size_t)l * F, h1, M, F, E);
    // h @ W2^T + c2 -> tmp f32; x = LN(x + tmp)
    gemm_bf16_kernel<false, false><<<dim3(E / 128, M / 128), 256, 0, stream>>>(
        h1, wl2b + (size_t)l * E * F, l2B + (size_t)l * E, tmp, M, E, F);
    add_ln_kernel<<<M, 256, 0, stream>>>(xf, tmp, ln2w + (size_t)l * E, ln2b + (size_t)l * E,
                                         xf, xb, E);
  }

  // LM head: out = x @ headW^T + head_b -> f32 [M, V]
  gemm_bf16_kernel<false, false><<<dim3(V / 128, M / 128), 256, 0, stream>>>(
      xb, wheadb, headB, out, M, V, E);
}
